// MViT_85280870630118
// MI455X (gfx1250) — matrix operand order repaired and checked by reading, not compiled, not run on hardware
//
#include <hip/hip_runtime.h>
#include <hip/hip_bf16.h>
#include <stdint.h>

// ---------------- problem constants ----------------
#define TT 8
#define HH 28
#define WW 28
#define NSP (TT*HH*WW)          // 6272
#define NTOK (NSP + 1)          // 6273
#define DIMX 192
#define HEADS 2
#define HDIM 96
#define NPAD 6336               // attention row padding: multiple of 64
#define NPADX 6288              // GEMM row padding: 393*16
#define SCALE 0.10206207261596575f  // 96^-0.5
#define KT 64                   // keys per attention tile

typedef __attribute__((ext_vector_type(16))) _Float16 v16h;
typedef __attribute__((ext_vector_type(8)))  float    v8f;
typedef __attribute__((ext_vector_type(4)))  unsigned int v4u;
typedef __attribute__((ext_vector_type(8)))  int      v8i;
typedef __attribute__((ext_vector_type(4)))  int      v4i;

#if defined(__has_builtin)
#if __has_builtin(__builtin_amdgcn_tensor_load_to_lds) && __has_builtin(__builtin_amdgcn_s_wait_tensorcnt)
#define USE_TDM 1
#endif
#endif

__device__ __forceinline__ v8f wmma16(const v16h a, const v16h b, const v8f c) {
  return __builtin_amdgcn_wmma_f32_16x16x32_f16(false, a, false, b, (short)0, c, false, false);
}

__device__ __forceinline__ v8f zero8() {
  v8f z;
#pragma unroll
  for (int i = 0; i < 8; ++i) z[i] = 0.f;
  return z;
}

union h16u { uint4 u[2]; v16h h; };

// ---- A operand (16x32 f16): lane<16 -> M=lane, Koff 0; lane>=16 -> M=lane-16, Koff 8
// element e -> K = kbase + ((e&8)?16:0) + ko + (e&7). (kbase+ko)*2 % 16 == 0 and 16B rows
// -> two b128 loads per lane.
__device__ __forceinline__ v16h load_a_vec(const _Float16* base, int ld, int rbase, int kbase) {
  const int lane = threadIdx.x & 31;
  const int r  = rbase + (lane & 15);
  const int ko = (lane & 16) ? 8 : 0;
  const _Float16* p = base + (size_t)r * ld + kbase + ko;
  h16u u;
  u.u[0] = *(const uint4*)p;
  u.u[1] = *(const uint4*)(p + 16);
  return u.h;
}

// ---- B operand (32x16): lane -> N = nbase + (lane&15), Koff = (lane&16)?8:0;
// element e -> K = kbase + ((e&8)?16:0) + Koff + (e&7). Memory row-major by K -> strided scalar loads.
__device__ __forceinline__ v16h load_b_vec(const _Float16* base, int ld, int kbase, int nbase) {
  const int lane = threadIdx.x & 31;
  const int ko = (lane & 16) ? 8 : 0;
  const _Float16* p = base + (size_t)(kbase + ko) * ld + nbase + (lane & 15);
  v16h v;
#pragma unroll
  for (int e = 0; e < 16; ++e) v[e] = p[(size_t)(((e & 8) ? 16 : 0) + (e & 7)) * ld];
  return v;
}

// ================= prep kernels: pack to f16 =================
__global__ void pack_x_kernel(const float* __restrict__ x, _Float16* __restrict__ xh) {
  const size_t i = (size_t)blockIdx.x * blockDim.x + threadIdx.x;
  if (i >= (size_t)NPADX * DIMX) return;
  const size_t row = i / DIMX;
  xh[i] = (row < NTOK) ? (_Float16)x[i] : (_Float16)0.f;
}

__global__ void pack_w_kernel(const float* __restrict__ qkvw, const float* __restrict__ projw,
                              _Float16* __restrict__ wqkvt, _Float16* __restrict__ wprojt) {
  const size_t i = (size_t)blockIdx.x * blockDim.x + threadIdx.x;
  const size_t n_qkv = (size_t)DIMX * 3 * DIMX;      // 192*576 (transposed)
  if (i < n_qkv) {
    const int k = (int)(i / (3 * DIMX)), n = (int)(i % (3 * DIMX));
    wqkvt[i] = (_Float16)qkvw[(size_t)n * DIMX + k];
  } else if (i < n_qkv + (size_t)DIMX * DIMX) {
    const size_t j = i - n_qkv;
    const int k = (int)(j / DIMX), n = (int)(j % DIMX);
    wprojt[j] = (_Float16)projw[(size_t)n * DIMX + k];
  }
}

// ================= Kernel 1: QKV projection (WMMA, all-f16 operands) =================
__global__ __launch_bounds__(32) void qkv_gemm_kernel(
    const _Float16* __restrict__ xh, const _Float16* __restrict__ wqkvt,
    const float* __restrict__ b,
    float* __restrict__ qr, float* __restrict__ kr, float* __restrict__ vr)
{
  const int rbase = blockIdx.x * 16;
  const int fbase = blockIdx.y * 16;
  const int lane = threadIdx.x & 31;
  v8f acc = zero8();
#pragma unroll
  for (int kc = 0; kc < DIMX; kc += 32)
    acc = wmma16(load_a_vec(xh, DIMX, rbase, kc),
                 load_b_vec(wqkvt, 3 * DIMX, kc, fbase), acc);
  const int f  = fbase + (lane & 15);
  const int which = f / DIMX;
  const int head  = (f % DIMX) / HDIM;
  const int ch    = f % HDIM;
  float* dst = (which == 0) ? qr : (which == 1) ? kr : vr;
  const float bias = b[f];
#pragma unroll
  for (int i = 0; i < 8; ++i) {
    const int row = rbase + i + ((lane & 16) ? 8 : 0);
    if (row < NTOK)
      dst[((size_t)head * NTOK + row) * HDIM + ch] = acc[i] + bias;
  }
}

// ================= Kernel 2: depthwise 3x3x3 conv + LayerNorm =================
__global__ __launch_bounds__(96) void pool_ln_kernel(
    const float* __restrict__ qr, const float* __restrict__ kr, const float* __restrict__ vr,
    const float* __restrict__ wq, const float* __restrict__ wk, const float* __restrict__ wv,
    const float* __restrict__ gq, const float* __restrict__ bq,
    const float* __restrict__ gk, const float* __restrict__ bk,
    const float* __restrict__ gv, const float* __restrict__ bv,
    float* __restrict__ qo, float* __restrict__ ko, float* __restrict__ vo)
{
  const int which = blockIdx.z;
  const float* raw = (which == 0) ? qr : (which == 1) ? kr : vr;
  const float* wc  = (which == 0) ? wq : (which == 1) ? wk : wv;
  const float* g   = (which == 0) ? gq : (which == 1) ? gk : gv;
  const float* be  = (which == 0) ? bq : (which == 1) ? bk : bv;
  float* out       = (which == 0) ? qo : (which == 1) ? ko : vo;

  const int head = blockIdx.y;
  const int tok  = blockIdx.x;
  const int ch   = threadIdx.x;
  const float* src = raw + (size_t)head * NTOK * HDIM;

  float val;
  if (tok == 0) {
    val = src[ch];
  } else {
    const int s = tok - 1;
    const int t = s / (HH * WW);
    const int h = (s / WW) % HH;
    const int w = s % WW;
    float acc = 0.f;
#pragma unroll
    for (int dt = -1; dt <= 1; ++dt)
#pragma unroll
      for (int dh = -1; dh <= 1; ++dh)
#pragma unroll
        for (int dw = -1; dw <= 1; ++dw) {
          const int t2 = t + dt, h2 = h + dh, w2 = w + dw;
          if (t2 < 0 || t2 >= TT || h2 < 0 || h2 >= HH || w2 < 0 || w2 >= WW) continue;
          const int ns = (t2 * HH + h2) * WW + w2;
          acc += src[(size_t)(1 + ns) * HDIM + ch] *
                 wc[ch * 27 + (dt + 1) * 9 + (dh + 1) * 3 + (dw + 1)];
        }
    val = acc;
  }
  __shared__ float red[6];
  float s1 = val, s2 = val * val;
#pragma unroll
  for (int off = 1; off < 32; off <<= 1) {
    s1 += __shfl_xor(s1, off, 32);
    s2 += __shfl_xor(s2, off, 32);
  }
  const int wid = ch >> 5;
  if ((ch & 31) == 0) { red[wid] = s1; red[3 + wid] = s2; }
  __syncthreads();
  const float mean = (red[0] + red[1] + red[2]) * (1.f / 96.f);
  const float var  = (red[3] + red[4] + red[5]) * (1.f / 96.f) - mean * mean;
  const float rstd = rsqrtf(var + 1e-5f);
  out[((size_t)head * NTOK + tok) * HDIM + ch] = (val - mean) * rstd * g[ch] + be[ch];
}

// ================= Kernel 3: decomposed rel-pos bias tables =================
__global__ __launch_bounds__(64) void relbias_kernel(
    const float* __restrict__ q_pool, const float* __restrict__ rel_t,
    const float* __restrict__ rel_h, const float* __restrict__ rel_w,
    float* __restrict__ biasq)
{
  const int head = blockIdx.y;
  const int s = blockIdx.x;
  const int j = threadIdx.x;
  const int t = s / (HH * WW);
  const int h = (s / WW) % HH;
  const int w = s % WW;
  const float* qv = q_pool + ((size_t)head * NTOK + (s + 1)) * HDIM;
  const float* rp; int idx;
  if (j < TT)            { rp = rel_t; idx = t - j + (TT - 1); }
  else if (j < TT + HH)  { rp = rel_h; idx = h - (j - TT) + (HH - 1); }
  else                   { rp = rel_w; idx = w - (j - TT - HH) + (WW - 1); }
  const float* rv = rp + (size_t)idx * HDIM;
  float acc = 0.f;
#pragma unroll 8
  for (int c = 0; c < HDIM; ++c) acc += qv[c] * rv[c];
  biasq[((size_t)head * NSP + s) * 64 + j] = acc;
}

// ================= Kernel 4: pack pooled q/k/v to f16, padded =================
__global__ void pack_qkv_kernel(
    const float* __restrict__ qp, const float* __restrict__ kp, const float* __restrict__ vp,
    _Float16* __restrict__ qh, _Float16* __restrict__ kh, _Float16* __restrict__ vh)
{
  const size_t i = (size_t)blockIdx.x * blockDim.x + threadIdx.x;
  const size_t total = (size_t)HEADS * NPAD * HDIM;
  if (i >= total) return;
  const size_t head = i / ((size_t)NPAD * HDIM);
  const size_t rem  = i % ((size_t)NPAD * HDIM);
  const size_t row  = rem / HDIM, ch = rem % HDIM;
  float q = 0.f, k = 0.f, v = 0.f;
  if (row < NTOK) {
    const size_t src = (head * NTOK + row) * HDIM + ch;
    q = qp[src]; k = kp[src]; v = vp[src];
  }
  qh[i] = (_Float16)q; kh[i] = (_Float16)k; vh[i] = (_Float16)v;
}

// ================= Kernel 5: fused flash attention =================
// grid (NPAD/64, HEADS), block 128 (4 waves x 16 query rows); 64-key tiles
#define KLD 72      // ktile_t row stride (halfs): 144B, 16B aligned
#define VLD 96      // vtile row stride (halfs): 192B
#define PLDK 72     // ptile row stride
__global__ __launch_bounds__(128) void attn_flash_kernel(
    const _Float16* __restrict__ qh, const _Float16* __restrict__ kh,
    const _Float16* __restrict__ vh, const float* __restrict__ biasq,
    const float* __restrict__ q_pool, _Float16* __restrict__ attn_h)
{
  const int head = blockIdx.y;
  const int qwg  = blockIdx.x * 64;
  const int tid  = threadIdx.x;
  const int wave = tid >> 5;
  const int lane = tid & 31;

  const _Float16* Q  = qh + (size_t)head * NPAD * HDIM;
  const _Float16* Kg = kh + (size_t)head * NPAD * HDIM;
  const _Float16* Vg = vh + (size_t)head * NPAD * HDIM;

  __shared__ __align__(16) _Float16 ktile_t[HDIM][KLD];   // transposed: [ch][key]
  __shared__ __align__(16) _Float16 vtile[KT][VLD];       // [key][ch]
  __shared__ __align__(16) _Float16 ptile[4][16][PLDK];
  __shared__ float btab[64][64];

  for (int idx = tid; idx < 64 * 64; idx += 128) {
    const int ql = idx >> 6, j = idx & 63;
    const int tok = qwg + ql;
    float b = 0.f;
    if (tok >= 1 && tok < NTOK) b = biasq[((size_t)head * NSP + (tok - 1)) * 64 + j];
    btab[ql][j] = b;
  }

  const int qrow = qwg + wave * 16;
  const v16h a0 = load_a_vec(Q, HDIM, qrow, 0);
  const v16h a1 = load_a_vec(Q, HDIM, qrow, 32);
  const v16h a2 = load_a_vec(Q, HDIM, qrow, 64);

  v8f O[6];
#pragma unroll
  for (int c = 0; c < 6; ++c) O[c] = zero8();
  float mrow[8], lrow[8];
#pragma unroll
  for (int i = 0; i < 8; ++i) { mrow[i] = -1e30f; lrow[i] = 0.f; }

  const int kcol0 = lane & 15;

  for (int kb = 0; kb < NPAD; kb += KT) {
    __syncthreads();

    // ---- K tile: transposed cooperative load (2 keys x 8 ch blocks) ----
    for (int blk = tid; blk < 32 * 12; blk += 128) {
      const int kp = (blk & 31) * 2;
      const int cb = (blk >> 5) * 8;
      const uint4 A  = *(const uint4*)(Kg + (size_t)(kb + kp) * HDIM + cb);
      const uint4 Bv = *(const uint4*)(Kg + (size_t)(kb + kp + 1) * HDIM + cb);
      const uint32_t aw[4] = {A.x, A.y, A.z, A.w};
      const uint32_t bw[4] = {Bv.x, Bv.y, Bv.z, Bv.w};
#pragma unroll
      for (int wq = 0; wq < 4; ++wq) {
        *(uint32_t*)&ktile_t[cb + 2 * wq][kp]     = (aw[wq] & 0xFFFFu) | (bw[wq] << 16);
        *(uint32_t*)&ktile_t[cb + 2 * wq + 1][kp] = (aw[wq] >> 16) | (bw[wq] & 0xFFFF0000u);
      }
    }

    // ---- V tile: Tensor Data Mover (async DMA) or manual fallback ----
#ifdef USE_TDM
    if (wave == 0) {
      const unsigned long long ga = (unsigned long long)(uintptr_t)(Vg + (size_t)kb * HDIM);
      const unsigned int ldsa = (unsigned int)(uintptr_t)&vtile[0][0];
      v4u g0; g0[0] = 1u; g0[1] = ldsa;
      g0[2] = (unsigned int)ga;
      g0[3] = ((unsigned int)(ga >> 32) & 0x01FFFFFFu) | (2u << 30);   // type=2 (image)
      v8i g1;                                   // 64x96 2-B tile, contiguous
      g1[0] = 0x10000;                          // data_size = 1 (2 bytes)
      g1[1] = (int)(96u << 16);                 // tensor_dim0 = 96
      g1[2] = (int)(64u << 16);                 // tensor_dim1 = 64
      g1[3] = (int)(96u << 16);                 // tile_dim0 = 96
      g1[4] = 64;                               // tile_dim1 = 64
      g1[5] = 96;                               // tensor_dim0_stride = 96
      g1[6] = 0; g1[7] = 0;
      v4i gz; gz[0] = gz[1] = gz[2] = gz[3] = 0;
#if defined(__clang_major__) && (__clang_major__ >= 23)
      v8i gz8;
#pragma unroll
      for (int z = 0; z < 8; ++z) gz8[z] = 0;
      __builtin_amdgcn_tensor_load_to_lds(g0, g1, gz, gz, gz8, 0);
#else
      __builtin_amdgcn_tensor_load_to_lds(g0, g1, gz, gz, 0);
#endif
      __builtin_amdgcn_s_wait_tensorcnt(0);
    }
#else
    for (int d = tid; d < KT * 12; d += 128) {
      const int r = d / 12, c = (d % 12) * 8;
      *(uint4*)&vtile[r][c] = *(const uint4*)(Vg + (size_t)(kb + r) * HDIM + c);
    }
#endif
    if (kb + KT < NPAD) {
      __builtin_prefetch(Kg + (size_t)(kb + KT) * HDIM, 0, 1);
      __builtin_prefetch(Vg + (size_t)(kb + KT) * HDIM, 0, 1);
    }
    __syncthreads();

    // ---- S = Q * K^T : 16x64 per wave, 12 WMMA ----
    v8f S[4];
#pragma unroll
    for (int j = 0; j < 4; ++j) S[j] = zero8();
#pragma unroll
    for (int kc = 0; kc < HDIM; kc += 32) {
      const v16h a = (kc == 0) ? a0 : (kc == 32) ? a1 : a2;
#pragma unroll
      for (int j = 0; j < 4; ++j)
        S[j] = wmma16(a, load_b_vec(&ktile_t[0][0], KLD, kc, j * 16), S[j]);
    }

    // ---- scale + rel bias + padding mask ----
    int itab[4], ihtab[4], iwtab[4];
    bool kval[4];
#pragma unroll
    for (int j = 0; j < 4; ++j) {
      const int kt = kb + kcol0 + j * 16;
      kval[j] = (kt < NTOK);
      itab[j] = -1;
      if (kt >= 1 && kt < NTOK) {
        const int s = kt - 1;
        itab[j] = s / (HH * WW);
        ihtab[j] = 8 + (s / WW) % HH;
        iwtab[j] = 36 + s % WW;
      }
    }
#pragma unroll
    for (int i = 0; i < 8; ++i) {
      const int qloc = wave * 16 + i + ((lane & 16) ? 8 : 0);
      const int qt = qwg + qloc;
      const bool qok = (qt >= 1 && qt < NTOK);
#pragma unroll
      for (int j = 0; j < 4; ++j) {
        float s = S[j][i] * SCALE;
        if (qok && itab[j] >= 0)
          s += btab[qloc][itab[j]] + btab[qloc][ihtab[j]] + btab[qloc][iwtab[j]];
        S[j][i] = kval[j] ? s : -1e30f;
      }
    }

    // ---- online softmax ----
#pragma unroll
    for (int i = 0; i < 8; ++i) {
      float mx = fmaxf(fmaxf(S[0][i], S[1][i]), fmaxf(S[2][i], S[3][i]));
#pragma unroll
      for (int off = 1; off < 16; off <<= 1) mx = fmaxf(mx, __shfl_xor(mx, off, 32));
      const float mnew = fmaxf(mrow[i], mx);
      const float corr = __expf(mrow[i] - mnew);
      mrow[i] = mnew;
      float rs = 0.f;
#pragma unroll
      for (int j = 0; j < 4; ++j) {
        const float p = __expf(S[j][i] - mnew);
        S[j][i] = p;
        rs += p;
      }
#pragma unroll
      for (int off = 1; off < 16; off <<= 1) rs += __shfl_xor(rs, off, 32);
      lrow[i] = lrow[i] * corr + rs;
#pragma unroll
      for (int c = 0; c < 6; ++c) O[c][i] *= corr;
    }

    // ---- stage P to LDS (C layout -> A layout) ----
#pragma unroll
    for (int i = 0; i < 8; ++i) {
      const int m = i + ((lane & 16) ? 8 : 0);
#pragma unroll
      for (int j = 0; j < 4; ++j)
        ptile[wave][m][kcol0 + 16 * j] = (_Float16)S[j][i];
    }
    __syncthreads();

    // ---- O += P * V : 12 WMMA ----
    const v16h ap0 = load_a_vec(&ptile[wave][0][0], PLDK, 0, 0);
    const v16h ap1 = load_a_vec(&ptile[wave][0][0], PLDK, 0, 32);
#pragma unroll
    for (int c = 0; c < 6; ++c) {
      O[c] = wmma16(ap0, load_b_vec(&vtile[0][0], VLD, 0,  c * 16), O[c]);
      O[c] = wmma16(ap1, load_b_vec(&vtile[0][0], VLD, 32, c * 16), O[c]);
    }
  }

  // ---- epilogue: normalize, residual, store f16 (input to proj GEMM) ----
#pragma unroll
  for (int c = 0; c < 6; ++c) {
    const int ch = c * 16 + kcol0;
#pragma unroll
    for (int i = 0; i < 8; ++i) {
      const int row = qwg + wave * 16 + i + ((lane & 16) ? 8 : 0);
      if (row < NPADX) {
        float val = 0.f;
        if (row < NTOK) {
          val = O[c][i] / lrow[i];
          if (row >= 1) val += q_pool[((size_t)head * NTOK + row) * HDIM + ch];
        }
        attn_h[(size_t)row * DIMX + head * HDIM + ch] = (_Float16)val;
      }
    }
  }
}

// ================= Kernel 6: output projection (WMMA, all-f16 operands) =================
__global__ __launch_bounds__(32) void proj_gemm_kernel(
    const _Float16* __restrict__ attn_h, const _Float16* __restrict__ wprojt,
    const float* __restrict__ b, float* __restrict__ out)
{
  const int rbase = blockIdx.x * 16;
  const int fbase = blockIdx.y * 16;
  const int lane = threadIdx.x & 31;
  v8f acc = zero8();
#pragma unroll
  for (int kc = 0; kc < DIMX; kc += 32)
    acc = wmma16(load_a_vec(attn_h, DIMX, rbase, kc),
                 load_b_vec(wprojt, DIMX, kc, fbase), acc);
  const int f = fbase + (lane & 15);
  const float bias = b[f];
#pragma unroll
  for (int i = 0; i < 8; ++i) {
    const int row = rbase + i + ((lane & 16) ? 8 : 0);
    if (row < NTOK) out[(size_t)row * DIMX + f] = acc[i] + bias;
  }
}

// ================= host launcher =================
extern "C" void kernel_launch(void* const* d_in, const int* in_sizes, int n_in,
                              void* d_out, int out_size, void* d_ws, size_t ws_size,
                              hipStream_t stream) {
  const float* x      = (const float*)d_in[0];
  const float* qkv_w  = (const float*)d_in[1];
  const float* qkv_b  = (const float*)d_in[2];
  const float* proj_w = (const float*)d_in[3];
  const float* proj_b = (const float*)d_in[4];
  const float* pqw    = (const float*)d_in[5];
  const float* pkw    = (const float*)d_in[6];
  const float* pvw    = (const float*)d_in[7];
  const float* nqg    = (const float*)d_in[8];
  const float* nqb    = (const float*)d_in[9];
  const float* nkg    = (const float*)d_in[10];
  const float* nkb    = (const float*)d_in[11];
  const float* nvg    = (const float*)d_in[12];
  const float* nvb    = (const float*)d_in[13];
  const float* rel_h  = (const float*)d_in[14];
  const float* rel_w  = (const float*)d_in[15];
  const float* rel_t  = (const float*)d_in[16];

  char* ws = (char*)d_ws;
  size_t off = 0;
  auto alloc = [&](size_t bytes) { size_t o = off; off += (bytes + 255) & ~(size_t)255; return o; };
  const size_t n_htc = (size_t)HEADS * NTOK * HDIM;
  const size_t n_pad = (size_t)HEADS * NPAD * HDIM;

  float* q_raw  = (float*)(ws + alloc(n_htc * 4));
  float* k_raw  = (float*)(ws + alloc(n_htc * 4));
  float* v_raw  = (float*)(ws + alloc(n_htc * 4));
  float* q_pool = (float*)(ws + alloc(n_htc * 4));
  float* k_pool = (float*)(ws + alloc(n_htc * 4));
  float* v_pool = (float*)(ws + alloc(n_htc * 4));
  _Float16* qh  = (_Float16*)(ws + alloc(n_pad * 2));
  _Float16* kh  = (_Float16*)(ws + alloc(n_pad * 2));
  _Float16* vh  = (_Float16*)(ws + alloc(n_pad * 2));
  float* biasq  = (float*)(ws + alloc((size_t)HEADS * NSP * 64 * 4));
  _Float16* xh     = (_Float16*)(ws + alloc((size_t)NPADX * DIMX * 2));
  _Float16* attn_h = (_Float16*)(ws + alloc((size_t)NPADX * DIMX * 2));
  _Float16* wqkvt  = (_Float16*)(ws + alloc((size_t)DIMX * 3 * DIMX * 2));
  _Float16* wprojt = (_Float16*)(ws + alloc((size_t)DIMX * DIMX * 2));

  const int row_tiles = (NTOK + 15) / 16;   // 393

  {
    const size_t nx = (size_t)NPADX * DIMX;
    pack_x_kernel<<<(unsigned)((nx + 255) / 256), 256, 0, stream>>>(x, xh);
    const size_t nw = (size_t)DIMX * 3 * DIMX + (size_t)DIMX * DIMX;
    pack_w_kernel<<<(unsigned)((nw + 255) / 256), 256, 0, stream>>>(qkv_w, proj_w, wqkvt, wprojt);
  }

  qkv_gemm_kernel<<<dim3(row_tiles, 36), 32, 0, stream>>>(xh, wqkvt, qkv_b, q_raw, k_raw, v_raw);

  pool_ln_kernel<<<dim3(NTOK, HEADS, 3), 96, 0, stream>>>(
      q_raw, k_raw, v_raw, pqw, pkw, pvw,
      nqg, nqb, nkg, nkb, nvg, nvb, q_pool, k_pool, v_pool);

  relbias_kernel<<<dim3(NSP, HEADS), 64, 0, stream>>>(q_pool, rel_t, rel_h, rel_w, biasq);

  {
    const size_t total = (size_t)HEADS * NPAD * HDIM;
    pack_qkv_kernel<<<(unsigned)((total + 255) / 256), 256, 0, stream>>>(
        q_pool, k_pool, v_pool, qh, kh, vh);
  }

  attn_flash_kernel<<<dim3(NPAD / 64, HEADS), 128, 0, stream>>>(
      qh, kh, vh, biasq, q_pool, attn_h);

  proj_gemm_kernel<<<dim3(row_tiles, 12), 32, 0, stream>>>(attn_h, wprojt, proj_b, (float*)d_out);
}
